// LongBertSelfAttention_84628035601123
// MI455X (gfx1250) — matrix operand order repaired and checked by reading, not compiled, not run on hardware
//
#include <hip/hip_runtime.h>
#include <hip/hip_bf16.h>

typedef __attribute__((ext_vector_type(16))) __bf16 v16bf;
typedef __attribute__((ext_vector_type(8)))  float  v8f;
typedef __attribute__((ext_vector_type(4)))  unsigned int v4u;
typedef __attribute__((ext_vector_type(8)))  int v8i;
typedef __attribute__((ext_vector_type(4)))  int v4i;

namespace {
constexpr int S_LEN   = 4096;
constexpr int D_MODEL = 768;
constexpr int NHEAD   = 12;
constexpr int DHEAD   = 64;
constexpr int WIN     = 256;   // one-sided window
constexpr int BATCH   = 2;
constexpr int M_TOT   = BATCH * S_LEN;          // 8192

constexpr size_t HID_ELEMS = (size_t)M_TOT * D_MODEL;        // 6291456
constexpr size_t W_ELEMS   = (size_t)D_MODEL * D_MODEL;      // 589824
constexpr size_t QKV_ELEMS = (size_t)BATCH * NHEAD * S_LEN * DHEAD; // 6291456
constexpr size_t HID_OFF = 0;
constexpr size_t W_OFF   = HID_OFF + HID_ELEMS;
constexpr size_t Q_OFF   = W_OFF + 3 * W_ELEMS;
constexpr size_t KT_OFF  = Q_OFF + QKV_ELEMS;
constexpr size_t V_OFF   = KT_OFF + QKV_ELEMS;
constexpr float  LOG2E   = 1.44269504088896340736f;
}

// ---- fragment loaders matching CDNA5 WMMA VGPR layouts (CDNA5 ISA §7.12.2) ----
__device__ __forceinline__ v16bf load_a16x32(const __bf16* base, int ld, int lane) {
    int m = lane & 15, kh = lane >> 4;
    const __bf16* p = base + (size_t)m * ld + kh * 8;
    union { v16bf v; uint4 q[2]; } u;
    u.q[0] = *reinterpret_cast<const uint4*>(p);
    u.q[1] = *reinterpret_cast<const uint4*>(p + 16);
    return u.v;
}
__device__ __forceinline__ v16bf load_b32x16(const __bf16* base, int ld, int lane) {
    int n = lane & 15, kh = lane >> 4;
    v16bf v;
    #pragma unroll
    for (int s = 0; s < 16; ++s) {
        int k = (s < 8) ? (kh * 8 + s) : (16 + kh * 8 + (s - 8));
        v[s] = base[(size_t)k * ld + n];
    }
    return v;
}

// ---- TDM: build D# per CDNA5 ISA and issue TENSOR_LOAD_TO_LDS ----
// 2D tile: tile_dim0 elements per row (contiguous), tile_dim1 rows, row stride stride0
// (in data_size=2B units). lds = byte offset of destination, ga = global byte address.
__device__ __forceinline__ void tdm_load_2d(unsigned long long ga, unsigned lds,
                                            unsigned tile0, unsigned tile1,
                                            unsigned long long stride0) {
    v4u g0;
    g0.x = 1u;                                                  // count=1, user mode
    g0.y = lds;                                                 // lds_addr [63:32]
    g0.z = (unsigned)(ga & 0xFFFFFFFFull);                      // global_addr lo
    g0.w = (unsigned)((ga >> 32) & 0x01FFFFFFull) | (2u << 30); // global_addr hi | type=2
    const unsigned td0 = 1u << 20, td1 = 1u << 20;              // huge => no OOB clipping
    v8i g1;
    g1[0] = (int)(1u << 16);                                    // data_size=1 -> 2 bytes
    g1[1] = (int)((td0 & 0xFFFFu) << 16);                       // tensor_dim0 lo16 @ [63:48]
    g1[2] = (int)((td0 >> 16) | ((td1 & 0xFFFFu) << 16));       // tensor_dim0 hi / dim1 lo
    g1[3] = (int)((td1 >> 16) | (tile0 << 16));                 // tensor_dim1 hi | tile_dim0
    g1[4] = (int)(tile1);                                       // tile_dim1 | tile_dim2=0
    g1[5] = (int)(stride0 & 0xFFFFFFFFull);                     // tensor_dim0_stride lo32
    g1[6] = (int)((stride0 >> 32) & 0xFFFFull);                 // stride0 hi16 | stride1 lo=0
    g1[7] = 0;                                                  // stride1 hi=0
    v4i z = {0, 0, 0, 0};
    v8i z8 = {0, 0, 0, 0, 0, 0, 0, 0};
    __builtin_amdgcn_tensor_load_to_lds(g0, g1, z, z, z8, 0);   // 6-arg form (this toolchain)
}

// ---- kernel 0: fp32 -> bf16 for hidden + the three weight matrices ----
__global__ __launch_bounds__(256) void cvt_kernel(const float* __restrict__ hid,
                                                  const float* __restrict__ wq,
                                                  const float* __restrict__ wk,
                                                  const float* __restrict__ wv,
                                                  __bf16* __restrict__ ws) {
    size_t idx = ((size_t)blockIdx.x * blockDim.x + threadIdx.x) * 4;
    const float* src;
    __bf16* dst;
    size_t off;
    if (idx < HID_ELEMS) {
        src = hid; dst = ws + HID_OFF; off = idx;
    } else {
        size_t t = idx - HID_ELEMS;
        int p = (int)(t / W_ELEMS);
        off = t - (size_t)p * W_ELEMS;
        src = (p == 0) ? wq : (p == 1) ? wk : wv;
        dst = ws + W_OFF + (size_t)p * W_ELEMS;
    }
    float4 f = *reinterpret_cast<const float4*>(src + off);
    dst[off + 0] = (__bf16)f.x;
    dst[off + 1] = (__bf16)f.y;
    dst[off + 2] = (__bf16)f.z;
    dst[off + 3] = (__bf16)f.w;
}

// ---- kernel 1: QKV projection GEMM via v_wmma_f32_16x16x32_bf16 ----
// grid = (M/256, N/64, 3). 8 waves/block; wave owns 32(M) x 64(N): two A-frags
// share each B-frag -> 8 WMMAs per 4 B-loads.
__global__ __launch_bounds__(256) void qkv_gemm(const __bf16* __restrict__ hid,
                                                const __bf16* __restrict__ wts,
                                                const float* __restrict__ bq,
                                                const float* __restrict__ bk,
                                                const float* __restrict__ bvv,
                                                __bf16* __restrict__ qout,
                                                __bf16* __restrict__ ktout,
                                                __bf16* __restrict__ vout) {
    const int lane = threadIdx.x & 31, wave = threadIdx.x >> 5;
    const int m0 = blockIdx.x * 256 + wave * 32;
    const int n0 = blockIdx.y * 64;
    const int proj = blockIdx.z;

    const __bf16* A = hid + (size_t)m0 * D_MODEL;
    const __bf16* Bw = wts + (size_t)proj * W_ELEMS;

    v8f c0[4], c1[4];
    #pragma unroll
    for (int j = 0; j < 4; ++j) {
        c0[j] = (v8f){0.f,0.f,0.f,0.f,0.f,0.f,0.f,0.f};
        c1[j] = (v8f){0.f,0.f,0.f,0.f,0.f,0.f,0.f,0.f};
    }

    for (int k0 = 0; k0 < D_MODEL; k0 += 32) {
        v16bf a0 = load_a16x32(A + k0, D_MODEL, lane);
        v16bf a1 = load_a16x32(A + (size_t)16 * D_MODEL + k0, D_MODEL, lane);
        #pragma unroll
        for (int j = 0; j < 4; ++j) {
            v16bf b = load_b32x16(Bw + (size_t)k0 * D_MODEL + n0 + 16 * j, D_MODEL, lane);
            c0[j] = __builtin_amdgcn_wmma_f32_16x16x32_bf16(false, a0, false, b,
                                                            (short)0, c0[j], false, false);
            c1[j] = __builtin_amdgcn_wmma_f32_16x16x32_bf16(false, a1, false, b,
                                                            (short)0, c1[j], false, false);
        }
    }

    const float* bias = (proj == 0) ? bq : (proj == 1) ? bk : bvv;
    const float scale = (proj == 0) ? 0.125f : 1.0f;   // 1/sqrt(DH)
    const int ln = lane & 15, hl = lane >> 4;
    #pragma unroll
    for (int j = 0; j < 4; ++j) {
        int n = n0 + j * 16 + ln;
        int h = n >> 6, d = n & 63;
        float bia = bias[n];
        #pragma unroll
        for (int half = 0; half < 2; ++half) {
            const v8f& c = half ? c1[j] : c0[j];
            #pragma unroll
            for (int r = 0; r < 8; ++r) {
                int m = m0 + half * 16 + r + hl * 8;
                int bidx = m >> 12, s = m & (S_LEN - 1);
                float val = (c[r] + bia) * scale;
                size_t head = (size_t)bidx * NHEAD + h;
                if (proj == 0)      qout [(head * S_LEN + s) * DHEAD + d] = (__bf16)val;
                else if (proj == 1) ktout[(head * DHEAD + d) * S_LEN + s] = (__bf16)val;
                else                vout [(head * S_LEN + s) * DHEAD + d] = (__bf16)val;
            }
        }
    }
}

// ---- kernel 2: banded flash attention, TDM-staged K/V, double-buffered ----
// grid = (S/64, H, B); 4 waves/block, each wave owns a 16-query strip.
__global__ __launch_bounds__(128) void attn_kernel(const __bf16* __restrict__ qb,
                                                   const __bf16* __restrict__ ktb,
                                                   const __bf16* __restrict__ vb,
                                                   const float* __restrict__ amask,
                                                   float* __restrict__ out) {
    __shared__ __align__(16) __bf16 ldsK[2][DHEAD * 32];   // K^T tile: 64 dims x 32 keys
    __shared__ __align__(16) __bf16 ldsV[2][32 * DHEAD];   // V tile:   32 keys x 64 dims
    __shared__ __align__(16) __bf16 ldsP[4][16 * 32];      // per-wave P bounce

    const int lane = threadIdx.x & 31, wave = threadIdx.x >> 5;
    const int ln = lane & 15, hl = lane >> 4;
    const int b = blockIdx.z, h = blockIdx.y;
    const int qb0 = blockIdx.x * 64;
    const int q0 = qb0 + wave * 16;

    const size_t head = (size_t)b * NHEAD + h;
    const __bf16* qh  = qb  + head * S_LEN * DHEAD;
    const __bf16* kth = ktb + head * DHEAD * S_LEN;
    const __bf16* vh  = vb  + head * S_LEN * DHEAD;
    const float*  am  = amask + (size_t)b * S_LEN;

    // block-uniform key band, clipped to [0,S): multiples of 32
    int jlo = qb0 - WIN;       if (jlo < 0) jlo = 0;
    int jhi = qb0 + 64 + WIN;  if (jhi > S_LEN) jhi = S_LEN;
    const int nt = (jhi - jlo) >> 5;

    // Q strip as two A fragments (DH=64 -> 2x K=32)
    v16bf aQ0 = load_a16x32(qh + (size_t)q0 * DHEAD,      DHEAD, lane);
    v16bf aQ1 = load_a16x32(qh + (size_t)q0 * DHEAD + 32, DHEAD, lane);

    v8f o[4];
    #pragma unroll
    for (int j = 0; j < 4; ++j) o[j] = (v8f){0.f,0.f,0.f,0.f,0.f,0.f,0.f,0.f};
    float mrow[8], lrow[8];
    #pragma unroll
    for (int r = 0; r < 8; ++r) { mrow[r] = -3.0e38f; lrow[r] = 0.f; }

    // prologue: wave0 DMAs tile 0 into buffer 0 (2 TDM ops per tile, in-order)
    if (wave == 0) {
        int j0 = jlo;
        tdm_load_2d((unsigned long long)(uintptr_t)(kth + j0),
                    (unsigned)(uintptr_t)&ldsK[0][0], 32, DHEAD, S_LEN);
        tdm_load_2d((unsigned long long)(uintptr_t)(vh + (size_t)j0 * DHEAD),
                    (unsigned)(uintptr_t)&ldsV[0][0], 32 * DHEAD, 0, 32 * DHEAD);
    }

    for (int t = 0; t < nt; ++t) {
        const int cur = t & 1;
        const int j0 = jlo + t * 32;
        if (wave == 0) {
            if (t + 1 < nt) {
                int j1 = jlo + (t + 1) * 32;
                tdm_load_2d((unsigned long long)(uintptr_t)(kth + j1),
                            (unsigned)(uintptr_t)&ldsK[1 - cur][0], 32, DHEAD, S_LEN);
                tdm_load_2d((unsigned long long)(uintptr_t)(vh + (size_t)j1 * DHEAD),
                            (unsigned)(uintptr_t)&ldsV[1 - cur][0], 32 * DHEAD, 0, 32 * DHEAD);
                __builtin_amdgcn_s_wait_tensorcnt(2);   // tile t done, t+1 in flight
            } else {
                __builtin_amdgcn_s_wait_tensorcnt(0);
            }
        }
        __syncthreads();   // buffer `cur` ready for all waves

        // scores: 16 queries x 32 keys from LDS-staged K^T
        v8f s[2];
        #pragma unroll
        for (int nh = 0; nh < 2; ++nh) {
            v16bf bLo = load_b32x16(&ldsK[cur][0]       + 16 * nh, 32, lane); // dims 0-31
            v16bf bHi = load_b32x16(&ldsK[cur][32 * 32] + 16 * nh, 32, lane); // dims 32-63
            v8f acc = (v8f){0.f,0.f,0.f,0.f,0.f,0.f,0.f,0.f};
            acc = __builtin_amdgcn_wmma_f32_16x16x32_bf16(false, aQ0, false, bLo, (short)0, acc, false, false);
            acc = __builtin_amdgcn_wmma_f32_16x16x32_bf16(false, aQ1, false, bHi, (short)0, acc, false, false);
            s[nh] = acc;
        }

        // mask (|i-j|<=W and key not padded): lane -> key, vgpr r -> query row
        float sc[2][8];
        #pragma unroll
        for (int nh = 0; nh < 2; ++nh) {
            int kidx = j0 + 16 * nh + ln;
            bool kok = (am[kidx] >= 0.f);
            #pragma unroll
            for (int r = 0; r < 8; ++r) {
                int qi  = q0 + r + hl * 8;
                int rel = kidx - qi;
                float v = s[nh][r];
                sc[nh][r] = (kok && rel <= WIN && rel >= -WIN) ? v : -1.0e9f;
            }
        }

        // online softmax: row reductions across the 16-lane half-groups
        float p0[8], p1[8];
        #pragma unroll
        for (int r = 0; r < 8; ++r) {
            float tmax = fmaxf(sc[0][r], sc[1][r]);
            #pragma unroll
            for (int off = 1; off < 16; off <<= 1)
                tmax = fmaxf(tmax, __shfl_xor(tmax, off));
            float mnew  = fmaxf(mrow[r], tmax);
            float alpha = exp2f((mrow[r] - mnew) * LOG2E);
            mrow[r] = mnew;
            p0[r] = exp2f((sc[0][r] - mnew) * LOG2E);
            p1[r] = exp2f((sc[1][r] - mnew) * LOG2E);
            float rs = p0[r] + p1[r];
            #pragma unroll
            for (int off = 1; off < 16; off <<= 1)
                rs += __shfl_xor(rs, off);
            lrow[r] = lrow[r] * alpha + rs;
            #pragma unroll
            for (int j = 0; j < 4; ++j) o[j][r] *= alpha;
        }

        // P: C-frag (lane->key) -> A-frag (lane->row) via per-wave LDS bounce
        #pragma unroll
        for (int r = 0; r < 8; ++r) {
            int m = r + hl * 8;
            ldsP[wave][m * 32 + ln]      = (__bf16)p0[r];
            ldsP[wave][m * 32 + 16 + ln] = (__bf16)p1[r];
        }
        asm volatile("s_wait_dscnt 0" ::: "memory");  // DS in-order per wave
        union { v16bf v; uint4 q[2]; } up;
        const uint4* lp = reinterpret_cast<const uint4*>(&ldsP[wave][0]);
        up.q[0] = lp[ln * 4 + hl];
        up.q[1] = lp[ln * 4 + 2 + hl];
        v16bf aP = up.v;

        // O += P (16x32) x V (32 keys x 64 dims) from LDS-staged V
        #pragma unroll
        for (int j = 0; j < 4; ++j) {
            v16bf bV = load_b32x16(&ldsV[cur][16 * j], DHEAD, lane);
            o[j] = __builtin_amdgcn_wmma_f32_16x16x32_bf16(false, aP, false, bV,
                                                           (short)0, o[j], false, false);
        }
        __syncthreads();   // all waves done with buffer `cur` before TDM reuses it
    }

    // normalize + query-padding mask, write fp32 output [B,S,H*DH]
    float qfac[8];
    #pragma unroll
    for (int r = 0; r < 8; ++r) {
        int qi = q0 + r + hl * 8;
        float inv = (lrow[r] > 0.f) ? (1.0f / lrow[r]) : 0.f;
        qfac[r] = (am[qi] < 0.f) ? 0.f : inv;
    }
    #pragma unroll
    for (int j = 0; j < 4; ++j) {
        int d = j * 16 + ln;
        #pragma unroll
        for (int r = 0; r < 8; ++r) {
            int qi = q0 + r + hl * 8;
            out[((size_t)b * S_LEN + qi) * D_MODEL + h * DHEAD + d] = o[j][r] * qfac[r];
        }
    }
}

extern "C" void kernel_launch(void* const* d_in, const int* in_sizes, int n_in,
                              void* d_out, int out_size, void* d_ws, size_t ws_size,
                              hipStream_t stream) {
    const float* hid   = (const float*)d_in[0];
    const float* amask = (const float*)d_in[1];
    const float* Wq    = (const float*)d_in[2];
    const float* bq    = (const float*)d_in[3];
    const float* Wk    = (const float*)d_in[4];
    const float* bk    = (const float*)d_in[5];
    const float* Wv    = (const float*)d_in[6];
    const float* bv    = (const float*)d_in[7];
    __bf16* ws = (__bf16*)d_ws;
    float* out = (float*)d_out;

    const size_t total4 = (HID_ELEMS + 3 * W_ELEMS) / 4;
    cvt_kernel<<<dim3((unsigned)(total4 / 256)), dim3(256), 0, stream>>>(hid, Wq, Wk, Wv, ws);

    qkv_gemm<<<dim3(M_TOT / 256, D_MODEL / 64, 3), dim3(256), 0, stream>>>(
        ws + HID_OFF, ws + W_OFF, bq, bk, bv,
        ws + Q_OFF, ws + KT_OFF, ws + V_OFF);

    attn_kernel<<<dim3(S_LEN / 64, NHEAD, BATCH), dim3(128), 0, stream>>>(
        ws + Q_OFF, ws + KT_OFF, ws + V_OFF, amask, out);
}
